// MultiheadAttention_40570261078719
// MI455X (gfx1250) — matrix operand order repaired and checked by reading, not compiled, not run on hardware
//
#include <hip/hip_runtime.h>
#include <hip/hip_fp16.h>

typedef _Float16 v16h __attribute__((ext_vector_type(16)));
typedef _Float16 v8h  __attribute__((ext_vector_type(8)));
typedef float    v8f  __attribute__((ext_vector_type(8)));

#define DM   1024
#define SEQ  2048
#define NB   2
#define NH   16
#define HD   64
#define KT   64                 // attention k-tile
#define MROWS (NB * SEQ)        // 4096 total rows

// ---- fragment helpers ------------------------------------------------------

__device__ __forceinline__ void load8h_f32(const float* __restrict__ p, _Float16* d) {
    const float4 a = *(const float4*)p;
    const float4 b = *(const float4*)(p + 4);
    d[0] = (_Float16)a.x; d[1] = (_Float16)a.y; d[2] = (_Float16)a.z; d[3] = (_Float16)a.w;
    d[4] = (_Float16)b.x; d[5] = (_Float16)b.y; d[6] = (_Float16)b.z; d[7] = (_Float16)b.w;
}
__device__ __forceinline__ void load8h_f16(const _Float16* __restrict__ p, _Float16* d) {
    v8h v = *(const v8h*)p;
#pragma unroll
    for (int j = 0; j < 8; ++j) d[j] = v[j];
}
template <typename T>
__device__ __forceinline__ void load8h(const T* p, _Float16* d);
template <> __device__ __forceinline__ void load8h<float>(const float* p, _Float16* d)       { load8h_f32(p, d); }
template <> __device__ __forceinline__ void load8h<_Float16>(const _Float16* p, _Float16* d) { load8h_f16(p, d); }

// B fragment: 16 contiguous f16 (K = half*16 .. +15 for this lane)
__device__ __forceinline__ v16h frag16h(const _Float16* __restrict__ p) {
    v8h lo = *(const v8h*)p;
    v8h hi = *(const v8h*)(p + 8);
    v16h r;
#pragma unroll
    for (int j = 0; j < 8; ++j) { r[j] = lo[j]; r[j + 8] = hi[j]; }
    return r;
}

__device__ __forceinline__ v8f wmma_f16(v16h a, v16h b, v8f c) {
    return __builtin_amdgcn_wmma_f32_16x16x32_f16(false, a, false, b, (short)0, c, false, false);
}

// ---- DPP butterfly reductions within each 16-lane half (pure VALU) ---------
template <int CTRL>
__device__ __forceinline__ float dpp_mov(float x) {
    return __int_as_float(__builtin_amdgcn_update_dpp(
        __float_as_int(x), __float_as_int(x), CTRL, 0xF, 0xF, false));
}
__device__ __forceinline__ float red_max16(float x) {
    x = fmaxf(x, dpp_mov<0xB1>(x));   // quad_perm xor1
    x = fmaxf(x, dpp_mov<0x4E>(x));   // quad_perm xor2
    x = fmaxf(x, dpp_mov<0x141>(x));  // row_half_mirror (combine quads in 8)
    x = fmaxf(x, dpp_mov<0x140>(x));  // row_mirror (combine halves in 16)
    return x;
}
__device__ __forceinline__ float red_sum16(float x) {
    x += dpp_mov<0xB1>(x);
    x += dpp_mov<0x4E>(x);
    x += dpp_mov<0x141>(x);
    x += dpp_mov<0x140>(x);
    return x;
}

// ---- CDNA5 async copy global -> LDS (inline asm; tracked by ASYNCcnt) ------
__device__ __forceinline__ void async_load_b128(unsigned int lds_off, unsigned long long gaddr) {
    asm volatile("global_load_async_to_lds_b128 %0, %1, off"
                 :: "v"(lds_off), "v"(gaddr) : "memory");
}
__device__ __forceinline__ void wait_async0() {
    asm volatile("s_wait_asynccnt 0" ::: "memory");
}

// ---- f32 -> f16 conversion pre-pass ----------------------------------------
__global__ __launch_bounds__(256)
void cvt_f32_f16(const float* __restrict__ s, _Float16* __restrict__ d, int n) {
    const int i = (blockIdx.x * 256 + threadIdx.x) * 8;
    if (i + 8 <= n) {
        _Float16 t[8];
        load8h_f32(s + i, t);
        v8h o;
#pragma unroll
        for (int j = 0; j < 8; ++j) o[j] = t[j];
        *(v8h*)(d + i) = o;
    }
}

// ---- GEMM: Y[M,N] = X[M,K] @ W[N,K]^T + bias  (M=4096, K=N=1024, W in f16) -
// One wave computes a 16x64 output strip. grid = 512 blocks x 256 threads.
template <typename TIN, typename TOUT>
__global__ __launch_bounds__(256)
void gemm_xwt_bias(const TIN* __restrict__ X, const _Float16* __restrict__ W16,
                   const float* __restrict__ bias, TOUT* __restrict__ Y) {
    const int lane = threadIdx.x & 31;
    const int wid  = (blockIdx.x * 256 + threadIdx.x) >> 5;   // 0..4095
    const int half = lane >> 4;
    const int col  = lane & 15;
    const int m0 = (wid >> 4) << 4;   // 256 m-tiles of 16
    const int n0 = (wid & 15) << 6;   // 16 n-strips of 64

    v8f acc[4] = {};
    const TIN* xrow = X + (size_t)(m0 + col) * DM;

    for (int k0 = 0; k0 < DM; k0 += 32) {
        // A fragment: row = m0 + lane%16; K = k0 + half*16 .. +15 (contiguous)
        _Float16 t[16];
        load8h<TIN>(xrow + k0 + half * 16, t);
        load8h<TIN>(xrow + k0 + half * 16 + 8, t + 8);
        v16h a;
#pragma unroll
        for (int j = 0; j < 16; ++j) a[j] = t[j];

#pragma unroll
        for (int c = 0; c < 4; ++c) {
            const _Float16* wrow = W16 + (size_t)(n0 + c * 16 + col) * DM + k0 + half * 16;
            acc[c] = wmma_f16(a, frag16h(wrow), acc[c]);
        }
    }

#pragma unroll
    for (int c = 0; c < 4; ++c) {
        const float bv = bias[n0 + c * 16 + col];
#pragma unroll
        for (int i = 0; i < 8; ++i) {
            const int row = m0 + i + half * 8;
            Y[(size_t)row * DM + n0 + c * 16 + col] = (TOUT)(acc[c][i] + bv);
        }
    }
}

// ---- Flash attention -------------------------------------------------------
// grid = NB*NH*16 blocks of 256 threads; each wave owns 16 q-rows (128/block).
__global__ __launch_bounds__(256)
void attn_kernel(const _Float16* __restrict__ Q, const _Float16* __restrict__ K,
                 const _Float16* __restrict__ V, _Float16* __restrict__ O) {
    __shared__ _Float16 kT[KT * HD];       // K tile row-major [krow][hd], async-filled
    __shared__ _Float16 vT[HD][KT + 8];    // V tile transposed [hd][kk], padded stride
    __shared__ _Float16 pl[8][16][KT];     // per-wave P tile (q-row x k-col)

    const int tid  = threadIdx.x;
    const int lane = tid & 31;
    const int w    = tid >> 5;
    const int half = lane >> 4;
    const int col  = lane & 15;

    const int chunk = blockIdx.x & 15;
    const int bh    = blockIdx.x >> 4;
    const int b     = bh >> 4;
    const int h     = bh & 15;
    const size_t base = (size_t)b * SEQ * DM + (size_t)h * HD;  // + s*DM + hd
    const int s0 = chunk * 128 + w * 16;

    const unsigned int kT_lds = (unsigned int)(uintptr_t)(void*)&kT[0];

    // Q A-fragments for hd 0..31 and 32..63 (row = s0 + lane%16)
    v16h qa0, qa1;
    {
        const _Float16* qrow = Q + base + (size_t)(s0 + col) * DM;
        _Float16 t[16];
        load8h_f16(qrow + half * 16, t);
        load8h_f16(qrow + half * 16 + 8, t + 8);
#pragma unroll
        for (int j = 0; j < 16; ++j) qa0[j] = t[j];
        load8h_f16(qrow + 32 + half * 16, t);
        load8h_f16(qrow + 32 + half * 16 + 8, t + 8);
#pragma unroll
        for (int j = 0; j < 16; ++j) qa1[j] = t[j];
    }

    v8f oa[4] = {};
    float mstat[8], lstat[8];
#pragma unroll
    for (int i = 0; i < 8; ++i) { mstat[i] = -1e30f; lstat[i] = 0.0f; }

    for (int kb = 0; kb < SEQ; kb += KT) {
        // --- async copy K tile (64x64 f16 = 512 x 16B segments) into LDS ----
#pragma unroll
        for (int s = 0; s < 2; ++s) {
            const int seg = tid * 2 + s;            // 0..511
            const int r   = seg >> 3;               // krow 0..63
            const int off = (seg & 7) * 8;          // hd segment
            async_load_b128(kT_lds + seg * 16,
                            (unsigned long long)(uintptr_t)(K + base + (size_t)(kb + r) * DM + off));
        }
        // --- cooperative transposed V-tile: 64 rows x 64 hd -> vT[hd][kk] ---
        {
            const int r  = tid >> 2;                // kk 0..63
            const int sc = (tid & 3) * 16;          // hd segment
            v8h v0 = *(const v8h*)(V + base + (size_t)(kb + r) * DM + sc);
            v8h v1 = *(const v8h*)(V + base + (size_t)(kb + r) * DM + sc + 8);
#pragma unroll
            for (int j = 0; j < 8; ++j) { vT[sc + j][r] = v0[j]; vT[sc + 8 + j][r] = v1[j]; }
        }
        wait_async0();
        __syncthreads();

        // --- scores: 4 C tiles covering k-cols [kb + t*16, +16) -------------
        v8f sc4[4];
#pragma unroll
        for (int t = 0; t < 4; ++t) {
            v8f ct = {};
            const _Float16* krt = &kT[(t * 16 + col) * HD];
            ct = wmma_f16(qa0, frag16h(krt + half * 16), ct);
            ct = wmma_f16(qa1, frag16h(krt + 32 + half * 16), ct);
            sc4[t] = ct;
        }

        // --- online softmax (row stats via DPP within 16-lane halves) -------
        float alpha[8];
#pragma unroll
        for (int i = 0; i < 8; ++i) {
            float s0v = sc4[0][i] * 0.125f, s1v = sc4[1][i] * 0.125f;
            float s2v = sc4[2][i] * 0.125f, s3v = sc4[3][i] * 0.125f;
            float rmax = red_max16(fmaxf(fmaxf(s0v, s1v), fmaxf(s2v, s3v)));
            const float mnew = fmaxf(mstat[i], rmax);
            alpha[i] = __expf(mstat[i] - mnew);
            s0v = __expf(s0v - mnew); s1v = __expf(s1v - mnew);
            s2v = __expf(s2v - mnew); s3v = __expf(s3v - mnew);
            const float rs = red_sum16((s0v + s1v) + (s2v + s3v));
            lstat[i] = lstat[i] * alpha[i] + rs;
            mstat[i] = mnew;
            sc4[0][i] = s0v; sc4[1][i] = s1v; sc4[2][i] = s2v; sc4[3][i] = s3v;
        }
#pragma unroll
        for (int c = 0; c < 4; ++c)
#pragma unroll
            for (int i = 0; i < 8; ++i) oa[c][i] *= alpha[i];

        // --- P: C-layout -> LDS (row-major f16) -----------------------------
#pragma unroll
        for (int t = 0; t < 4; ++t)
#pragma unroll
            for (int i = 0; i < 8; ++i)
                pl[w][i + half * 8][t * 16 + col] = (_Float16)sc4[t][i];
        __syncthreads();

        // --- P A-fragments (row = lane%16, K over the 64 k-cols) ------------
        v16h pa0, pa1;
        {
            const _Float16* pr = &pl[w][col][0];
            _Float16 t[16];
            load8h_f16(pr + half * 16, t);
            load8h_f16(pr + half * 16 + 8, t + 8);
#pragma unroll
            for (int j = 0; j < 16; ++j) pa0[j] = t[j];
            load8h_f16(pr + 32 + half * 16, t);
            load8h_f16(pr + 32 + half * 16 + 8, t + 8);
#pragma unroll
            for (int j = 0; j < 16; ++j) pa1[j] = t[j];
        }

        // --- O += P @ V : 4 output 16x16 tiles across hd --------------------
#pragma unroll
        for (int c = 0; c < 4; ++c) {
            const _Float16* vr = &vT[c * 16 + col][0];
            oa[c] = wmma_f16(pa0, frag16h(vr + half * 16), oa[c]);
            oa[c] = wmma_f16(pa1, frag16h(vr + 32 + half * 16), oa[c]);
        }
        __syncthreads();   // protect kT/vT/pl before next iteration overwrites
    }

    // finalize and store to [B,S,D] (already the (0,2,1,3)-permuted layout)
#pragma unroll
    for (int i = 0; i < 8; ++i) {
        const float inv = 1.0f / lstat[i];
#pragma unroll
        for (int c = 0; c < 4; ++c) oa[c][i] *= inv;
    }
#pragma unroll
    for (int c = 0; c < 4; ++c)
#pragma unroll
        for (int i = 0; i < 8; ++i)
            O[base + (size_t)(s0 + i + half * 8) * DM + c * 16 + col] = (_Float16)oa[c][i];
}

// ---- host launch -----------------------------------------------------------
extern "C" void kernel_launch(void* const* d_in, const int* in_sizes, int n_in,
                              void* d_out, int out_size, void* d_ws, size_t ws_size,
                              hipStream_t stream) {
    (void)in_sizes; (void)n_in; (void)out_size; (void)ws_size;
    const float* query = (const float*)d_in[0];
    const float* key_  = (const float*)d_in[1];
    const float* value = (const float*)d_in[2];
    const float* Wq = (const float*)d_in[3];
    const float* bq = (const float*)d_in[4];
    const float* Wk = (const float*)d_in[5];
    const float* bk = (const float*)d_in[6];
    const float* Wv = (const float*)d_in[7];
    const float* bv = (const float*)d_in[8];
    const float* Wo = (const float*)d_in[9];
    const float* bo = (const float*)d_in[10];

    const size_t WSZ = (size_t)DM * DM;            // 1M elements per weight
    const size_t SD  = (size_t)MROWS * DM;         // 4M elements per activation
    _Float16* wq16 = (_Float16*)d_ws;
    _Float16* wk16 = wq16 + WSZ;
    _Float16* wv16 = wk16 + WSZ;
    _Float16* wo16 = wv16 + WSZ;
    _Float16* qf   = wo16 + WSZ;
    _Float16* kf   = qf + SD;
    _Float16* vf   = kf + SD;
    _Float16* af   = vf + SD;

    const dim3 blk(256);
    const dim3 cvt_grid((unsigned)(WSZ / (256 * 8)));    // 512
    cvt_f32_f16<<<cvt_grid, blk, 0, stream>>>(Wq, wq16, (int)WSZ);
    cvt_f32_f16<<<cvt_grid, blk, 0, stream>>>(Wk, wk16, (int)WSZ);
    cvt_f32_f16<<<cvt_grid, blk, 0, stream>>>(Wv, wv16, (int)WSZ);
    cvt_f32_f16<<<cvt_grid, blk, 0, stream>>>(Wo, wo16, (int)WSZ);

    const dim3 gemm_grid(512);                     // 4096 waves = 256 m-tiles * 16 strips
    gemm_xwt_bias<float, _Float16><<<gemm_grid, blk, 0, stream>>>(query, wq16, bq, qf);
    gemm_xwt_bias<float, _Float16><<<gemm_grid, blk, 0, stream>>>(key_,  wk16, bk, kf);
    gemm_xwt_bias<float, _Float16><<<gemm_grid, blk, 0, stream>>>(value, wv16, bv, vf);

    const dim3 attn_grid(NB * NH * 16);            // 512 blocks
    attn_kernel<<<attn_grid, blk, 0, stream>>>(qf, kf, vf, af);

    gemm_xwt_bias<_Float16, float><<<gemm_grid, blk, 0, stream>>>(af, wo16, bo, (float*)d_out);
}
